// GCN_8057358647624
// MI455X (gfx1250) — compile-verified, matrix operand order checked by reading, not run on hardware
//
#include <hip/hip_runtime.h>

#define F_IN   512
#define F_HID  256
#define F_CLS  40
#define F_CLSP 48

typedef __attribute__((ext_vector_type(16))) __bf16 v16bf;
typedef __attribute__((ext_vector_type(8)))  float  v8f;

union Frag16 { v16bf v; uint4 q[2]; };

__device__ __forceinline__ unsigned short f2bf(float f) {
    unsigned int u = __float_as_uint(f);
    u += 0x7FFFu + ((u >> 16) & 1u);          // round-to-nearest-even
    return (unsigned short)(u >> 16);
}

// Load one 16x32 bf16 WMMA fragment (A layout; B uses same layout from W^T).
// p points at &buf[row*ldk + k0 + kg] where kg = 8*(lane>=16).
__device__ __forceinline__ v16bf load_frag16(const unsigned short* p) {
    Frag16 f;
    f.q[0] = *(const uint4*)(p);        // K = kg .. kg+7
    f.q[1] = *(const uint4*)(p + 16);   // K = 16+kg .. 16+kg+7
    return f.v;
}

__global__ void k_zero(float* __restrict__ p, long long n) {
    long long i = (long long)blockIdx.x * blockDim.x + threadIdx.x;
    long long s = (long long)gridDim.x * blockDim.x;
    for (; i < n; i += s) p[i] = 0.f;
}

__global__ void k_cast_bf16(const float* __restrict__ in, unsigned short* __restrict__ out,
                            long long n) {
    long long i = (long long)blockIdx.x * blockDim.x + threadIdx.x;
    long long s = (long long)gridDim.x * blockDim.x;
    for (; i < n; i += s) out[i] = f2bf(in[i]);
}

// W[K,N] f32 (row-major) -> Wt[Npad,K] bf16 (row-major), zero-padded rows n in [N,Npad)
__global__ void k_transpose_cast(const float* __restrict__ W, unsigned short* __restrict__ Wt,
                                 int K, int N, int Npad) {
    long long total = (long long)Npad * K;
    long long i = (long long)blockIdx.x * blockDim.x + threadIdx.x;
    long long s = (long long)gridDim.x * blockDim.x;
    for (; i < total; i += s) {
        int n = (int)(i / K);
        int k = (int)(i % K);
        Wt[i] = (n < N) ? f2bf(W[(long long)k * N + n]) : (unsigned short)0;
    }
}

// C[M,256] = A[M,512]bf16 @ W1t[256,512]bf16^T, f32 accumulate.
// Block = 256 thr = 8 waves; block tile 128x128; wave tile 32x64 (2x4 WMMA tiles).
__global__ __launch_bounds__(256) void k_gemm_xw1(const unsigned short* __restrict__ A,
                                                  const unsigned short* __restrict__ Bt,
                                                  float* __restrict__ C, int M) {
    const int K = F_IN, N = F_HID;
    const int lane = threadIdx.x & 31;
    const int wave = threadIdx.x >> 5;
    const int wm = wave & 3, wn = wave >> 2;
    const int m0 = blockIdx.x * 128 + wm * 32;
    const int n0 = blockIdx.y * 128 + wn * 64;
    const int l15 = lane & 15;
    const int kg = (lane >> 4) << 3;

    v8f z = {};
    v8f acc[2][4];
#pragma unroll
    for (int i = 0; i < 2; i++)
#pragma unroll
        for (int j = 0; j < 4; j++) acc[i][j] = z;

    for (int k0 = 0; k0 < K; k0 += 32) {
        v16bf a[2], b[4];
#pragma unroll
        for (int mt = 0; mt < 2; mt++) {
            int row = m0 + mt * 16 + l15;
            if (row >= M) row = M - 1;                      // clamp, no EXEC divergence
            a[mt] = load_frag16(A + (long long)row * K + k0 + kg);
        }
#pragma unroll
        for (int nt = 0; nt < 4; nt++) {
            int col = n0 + nt * 16 + l15;
            b[nt] = load_frag16(Bt + (long long)col * K + k0 + kg);
        }
#pragma unroll
        for (int mt = 0; mt < 2; mt++)
#pragma unroll
            for (int nt = 0; nt < 4; nt++)
                acc[mt][nt] = __builtin_amdgcn_wmma_f32_16x16x32_bf16(
                    false, a[mt], false, b[nt], (short)0, acc[mt][nt], false, false);
    }

    // Epilogue: uniform branch; full tiles take the unguarded straight-line path.
    // Row for accumulator element r is (mbase + halfsel + r) -> constant r*N imm offsets.
    const int halfsel = (lane < 16) ? 0 : 8;
    if ((blockIdx.x + 1) * 128 <= (unsigned)M) {
#pragma unroll
        for (int mt = 0; mt < 2; mt++) {
            float* base0 = C + (long long)(m0 + mt * 16 + halfsel) * N + n0 + l15;
#pragma unroll
            for (int nt = 0; nt < 4; nt++) {
                float* base = base0 + nt * 16;
#pragma unroll
                for (int r = 0; r < 8; r++) base[r * N] = acc[mt][nt][r];
            }
        }
    } else {
#pragma unroll
        for (int mt = 0; mt < 2; mt++)
#pragma unroll
            for (int nt = 0; nt < 4; nt++) {
                int ncol = n0 + nt * 16 + l15;
#pragma unroll
                for (int r = 0; r < 8; r++) {
                    int mrow = m0 + mt * 16 + halfsel + r;
                    if (mrow < M) C[(long long)mrow * N + ncol] = acc[mt][nt][r];
                }
            }
    }
}

// C[M,40] = A[M,256]bf16 @ W2t[48,256]bf16^T. Block tile 128x48; wave = 16 rows x 3 tiles.
__global__ __launch_bounds__(256) void k_gemm_hw2(const unsigned short* __restrict__ A,
                                                  const unsigned short* __restrict__ Bt,
                                                  float* __restrict__ C, int M) {
    const int K = F_HID;
    const int lane = threadIdx.x & 31;
    const int wave = threadIdx.x >> 5;
    const int m0 = blockIdx.x * 128 + wave * 16;
    const int l15 = lane & 15;
    const int kg = (lane >> 4) << 3;

    v8f z = {};
    v8f acc[3];
#pragma unroll
    for (int i = 0; i < 3; i++) acc[i] = z;

    for (int k0 = 0; k0 < K; k0 += 32) {
        int row = m0 + l15;
        if (row >= M) row = M - 1;
        v16bf a = load_frag16(A + (long long)row * K + k0 + kg);
#pragma unroll
        for (int nt = 0; nt < 3; nt++) {
            v16bf b = load_frag16(Bt + (long long)(nt * 16 + l15) * K + k0 + kg);
            acc[nt] = __builtin_amdgcn_wmma_f32_16x16x32_bf16(
                false, a, false, b, (short)0, acc[nt], false, false);
        }
    }

    const int halfsel = (lane < 16) ? 0 : 8;
    if ((blockIdx.x + 1) * 128 <= (unsigned)M) {
        float* base0 = C + (long long)(m0 + halfsel) * F_CLS + l15;
#pragma unroll
        for (int nt = 0; nt < 3; nt++) {
            int ncol = nt * 16 + l15;
            if (ncol < F_CLS) {                 // lane guard only on the last tile
                float* base = base0 + nt * 16;
#pragma unroll
                for (int r = 0; r < 8; r++) base[r * F_CLS] = acc[nt][r];
            }
        }
    } else {
#pragma unroll
        for (int nt = 0; nt < 3; nt++) {
            int ncol = nt * 16 + l15;
#pragma unroll
            for (int r = 0; r < 8; r++) {
                int mrow = m0 + halfsel + r;
                if (mrow < M && ncol < F_CLS) C[(long long)mrow * F_CLS + ncol] = acc[nt][r];
            }
        }
    }
}

// SpMM, F=256: one wave per edge; lane handles 2 float4 (8 floats) -> L2 atomics.
__global__ __launch_bounds__(256) void k_spmm256(const int* __restrict__ row,
                                                 const int* __restrict__ col,
                                                 const float* __restrict__ val,
                                                 const float* __restrict__ H,
                                                 float* __restrict__ out, int E) {
    int e = blockIdx.x * 8 + (threadIdx.x >> 5);
    if (e >= E) return;
    int lane = threadIdx.x & 31;
    int r = row[e], c = col[e];
    float v = val[e];
    const float4* src = (const float4*)(H + (long long)c * F_HID);
    float* dst = out + (long long)r * F_HID;
    __builtin_prefetch(dst + lane * 8, 0, 3);   // global_prefetch_b8: warm dst row in L2
#pragma unroll
    for (int j = 0; j < 2; j++) {
        int idx = lane + j * 32;
        float4 m = src[idx];
        unsafeAtomicAdd(dst + idx * 4 + 0, v * m.x);
        unsafeAtomicAdd(dst + idx * 4 + 1, v * m.y);
        unsafeAtomicAdd(dst + idx * 4 + 2, v * m.z);
        unsafeAtomicAdd(dst + idx * 4 + 3, v * m.w);
    }
}

// SpMM, F=40: one wave per edge.
__global__ __launch_bounds__(256) void k_spmm40(const int* __restrict__ row,
                                                const int* __restrict__ col,
                                                const float* __restrict__ val,
                                                const float* __restrict__ H,
                                                float* __restrict__ out, int E) {
    int e = blockIdx.x * 8 + (threadIdx.x >> 5);
    if (e >= E) return;
    int lane = threadIdx.x & 31;
    int r = row[e], c = col[e];
    float v = val[e];
    const float* src = H + (long long)c * F_CLS;
    float* dst = out + (long long)r * F_CLS;
    __builtin_prefetch(dst + lane, 0, 3);
    unsafeAtomicAdd(dst + lane, v * src[lane]);
    if (lane < 8) unsafeAtomicAdd(dst + 32 + lane, v * src[32 + lane]);
}

// h = relu(AH1 + b1), cast to bf16 for layer-2 WMMA
__global__ void k_bias_relu_cast(const float* __restrict__ H, const float* __restrict__ b,
                                 unsigned short* __restrict__ out, long long n) {
    long long i = (long long)blockIdx.x * blockDim.x + threadIdx.x;
    long long s = (long long)gridDim.x * blockDim.x;
    for (; i < n; i += s) {
        float h = H[i] + b[(int)(i & (F_HID - 1))];
        out[i] = f2bf(fmaxf(h, 0.f));
    }
}

// out = log_softmax(AH2 + b2) over 40 classes; one thread per node.
__global__ void k_bias_logsoftmax(const float* __restrict__ H, const float* __restrict__ b,
                                  float* __restrict__ out, int M) {
    int i = blockIdx.x * blockDim.x + threadIdx.x;
    if (i >= M) return;
    const float* row = H + (long long)i * F_CLS;
    float v[F_CLS];
    float mx = -3.4e38f;
#pragma unroll
    for (int j = 0; j < F_CLS; j++) { v[j] = row[j] + b[j]; mx = fmaxf(mx, v[j]); }
    float s = 0.f;
#pragma unroll
    for (int j = 0; j < F_CLS; j++) s += __expf(v[j] - mx);
    float l = mx + __logf(s);
    float* o = out + (long long)i * F_CLS;
#pragma unroll
    for (int j = 0; j < F_CLS; j++) o[j] = v[j] - l;
}

extern "C" void kernel_launch(void* const* d_in, const int* in_sizes, int n_in,
                              void* d_out, int out_size, void* d_ws, size_t ws_size,
                              hipStream_t stream) {
    const float* x  = (const float*)d_in[0];
    const int* erow = (const int*)d_in[1];
    const int* ecol = (const int*)d_in[2];
    const float* ev = (const float*)d_in[3];
    const float* W1 = (const float*)d_in[4];
    const float* b1 = (const float*)d_in[5];
    const float* W2 = (const float*)d_in[6];
    const float* b2 = (const float*)d_in[7];

    const int M = in_sizes[0] / F_IN;   // 100000
    const int E = in_sizes[1];          // 3200000

    char* ws = (char*)d_ws;
    size_t off = 0;
    auto carve = [&](size_t bytes) -> void* {
        void* p = ws + off;
        off += (bytes + 255) & ~(size_t)255;
        return p;
    };
    unsigned short* xbf = (unsigned short*)carve((size_t)M * F_IN * 2);
    unsigned short* W1t = (unsigned short*)carve((size_t)F_HID * F_IN * 2);
    unsigned short* W2t = (unsigned short*)carve((size_t)F_CLSP * F_HID * 2);
    float* H1  = (float*)carve((size_t)M * F_HID * 4);
    float* AH1 = (float*)carve((size_t)M * F_HID * 4);
    unsigned short* Hbf = (unsigned short*)carve((size_t)M * F_HID * 2);
    float* H2  = (float*)carve((size_t)M * F_CLS * 4);
    float* AH2 = (float*)carve((size_t)M * F_CLS * 4);

    // Precision conversion + weight transpose (B fragments become contiguous loads)
    k_cast_bf16<<<2048, 256, 0, stream>>>(x, xbf, (long long)M * F_IN);
    k_transpose_cast<<<512, 256, 0, stream>>>(W1, W1t, F_IN, F_HID, F_HID);
    k_transpose_cast<<<64, 256, 0, stream>>>(W2, W2t, F_HID, F_CLS, F_CLSP);

    // Zero scatter-add accumulators (d_ws is poisoned)
    k_zero<<<2048, 256, 0, stream>>>(AH1, (long long)M * F_HID);
    k_zero<<<2048, 256, 0, stream>>>(AH2, (long long)M * F_CLS);

    // Layer 1: WMMA GEMM -> SpMM -> bias+relu (+bf16 cast)
    dim3 g1((M + 127) / 128, F_HID / 128);
    k_gemm_xw1<<<g1, 256, 0, stream>>>(xbf, W1t, H1, M);
    k_spmm256<<<(E + 7) / 8, 256, 0, stream>>>(erow, ecol, ev, H1, AH1, E);
    k_bias_relu_cast<<<2048, 256, 0, stream>>>(AH1, b1, Hbf, (long long)M * F_HID);

    // Layer 2: WMMA GEMM -> SpMM -> bias + log_softmax
    k_gemm_hw2<<<(M + 127) / 128, 256, 0, stream>>>(Hbf, W2t, H2, M);
    k_spmm40<<<(E + 7) / 8, 256, 0, stream>>>(erow, ecol, ev, H2, AH2, E);
    k_bias_logsoftmax<<<(M + 255) / 256, 256, 0, stream>>>(AH2, b2, (float*)d_out, M);
}
